// ExtractorToPMoE_41721312313659
// MI455X (gfx1250) — matrix operand order repaired and checked by reading, not compiled, not run on hardware
//
#include <hip/hip_runtime.h>
#include <cstdint>
#include <cstddef>

// Problem sizes (match reference)
#define B_TOK 4096
#define D_IN  1024
#define E_EXP 16
#define H_HID 2048
#define C_CLS 1000

typedef __attribute__((ext_vector_type(16))) __bf16 v16bf;
typedef __attribute__((ext_vector_type(8)))  float  v8f;

struct alignas(16) Q16 { unsigned int u[4]; };
struct alignas(8)  U2  { unsigned int a, b; };
union FragU { Q16 q[2]; v16bf v; };

#if __has_builtin(__builtin_amdgcn_global_load_async_to_lds_b128)
#define HAVE_ASYNC_LDS 1
typedef int v4i_t __attribute__((vector_size(16)));
typedef __attribute__((address_space(1))) v4i_t* gas_v4i;
typedef __attribute__((address_space(3))) v4i_t* las_v4i;
#else
#define HAVE_ASYNC_LDS 0
#endif

// 16-byte global -> LDS copy; async (ASYNCcnt) when available.
static __device__ __forceinline__ void copy16_g2l(const void* g, void* l) {
#if HAVE_ASYNC_LDS
  __builtin_amdgcn_global_load_async_to_lds_b128(
      (gas_v4i)g, (las_v4i)l, 0, 0);
#else
  *(Q16*)l = *(const Q16*)g;
#endif
}

static __device__ __forceinline__ void wait_async_lds() {
#if HAVE_ASYNC_LDS
#if __has_builtin(__builtin_amdgcn_s_wait_asynccnt)
  __builtin_amdgcn_s_wait_asynccnt(0);
#else
  asm volatile("s_wait_asynccnt 0" ::: "memory");
#endif
#endif
}

static __device__ __forceinline__ unsigned short f2bf(float f) {
  // round-to-nearest-even fp32 -> bf16
  unsigned int u = __float_as_uint(f);
  unsigned int r = u + 0x7FFFu + ((u >> 16) & 1u);
  return (unsigned short)(r >> 16);
}

static __device__ __forceinline__ unsigned int f2bf_pk(float lo, float hi) {
  return (unsigned int)f2bf(lo) | ((unsigned int)f2bf(hi) << 16);
}

static __device__ __forceinline__ v8f wmma_bf16(const FragU& a, const FragU& b, v8f c) {
  return __builtin_amdgcn_wmma_f32_16x16x32_bf16(
      false, a.v, false, b.v, (short)0, c, false, false);
}

// ---------------------------------------------------------------------------
// Kernel 1: gate = softmax(x@Wg + bg), top-2, bucket scatter.
// One wave (32 lanes) per token: e = lane&15, p = lane>>4.
// ---------------------------------------------------------------------------
__global__ __launch_bounds__(256)
void gate_topk_kernel(const float* __restrict__ x, const float* __restrict__ Wg,
                      const float* __restrict__ bg, float* __restrict__ wv,
                      int* __restrict__ counts, int* __restrict__ tok_list) {
  const int wid  = threadIdx.x >> 5;
  const int lane = threadIdx.x & 31;
  const int b    = blockIdx.x * 8 + wid;
  const int e    = lane & 15;
  const int p    = lane >> 4;

  float acc = 0.f;
  const float* xr = x + (size_t)b * D_IN;
  for (int d = p; d < D_IN; d += 2)
    acc += xr[d] * Wg[d * E_EXP + e];
  acc += __shfl_down(acc, 16);     // lanes 0..15 hold full dot for expert e
  acc += bg[e];

  // softmax over 16 lanes (lanes 16..31 compute garbage, unused)
  float m = acc;
  #pragma unroll
  for (int off = 8; off >= 1; off >>= 1) m = fmaxf(m, __shfl_xor(m, off));
  float ex = __expf(acc - m);
  float s  = ex;
  #pragma unroll
  for (int off = 8; off >= 1; off >>= 1) s += __shfl_xor(s, off);
  float w = ex / s;

  // serial top-2 scan (ties -> lowest index, matches lax.top_k)
  float b0 = -1.f, b1v = -1.f; int i0 = 0, i1 = 0;
  #pragma unroll
  for (int j = 0; j < 16; ++j) {
    float wj = __shfl(w, j);
    if (wj > b0)       { b1v = b0; i1 = i0; b0 = wj; i0 = j; }
    else if (wj > b1v) { b1v = wj; i1 = j; }
  }
  if (lane == 0) {
    wv[b * 2 + 0] = b0;
    wv[b * 2 + 1] = b1v;
    int p0 = atomicAdd(&counts[i0], 1);
    tok_list[i0 * B_TOK + p0] = (b << 1);
    int p1 = atomicAdd(&counts[i1], 1);
    tok_list[i1 * B_TOK + p1] = (b << 1) | 1;
  }
}

// ---------------------------------------------------------------------------
// Kernel 2: routed fc1.  h[b*2+slot, :] = x[b]@W1[e] + b1[e]   (bf16 out)
// grid = (H/128, E); block = 256 (8 waves as 4M x 2N).
// Tile: M=64 tokens x N=128 cols, K-chunks of 32.
// Staging: float4 global loads; A packed to 8B LDS stores.
// ---------------------------------------------------------------------------
__global__ __launch_bounds__(256)
void moe_fc1_kernel(const float* __restrict__ x, const float* __restrict__ W1,
                    const float* __restrict__ b1, const int* __restrict__ counts,
                    const int* __restrict__ tok_list,
                    unsigned short* __restrict__ hbuf) {
  const int e     = blockIdx.y;
  const int cBase = blockIdx.x * 128;
  const int tid   = threadIdx.x;
  const int wid   = tid >> 5;
  const int lane  = tid & 31;
  const int nt    = counts[e];
  const float* W1e = W1 + (size_t)e * D_IN * H_HID;

  __shared__ unsigned short ldsA[64 * 40];    // [m][kk], stride 40 halves
  __shared__ unsigned short ldsB[128 * 40];   // [col][kk], stride 40 halves
  __shared__ int s_tok[64];

  const int n   = lane & 15;
  const int hi  = lane >> 4;
  const int aK  = hi ? 8 : 0;    // A fragment K base (halves)
  const int bK  = hi ? 8 : 0;    // B fragment K base (halves)
  const int mw  = wid & 3;       // M strip (16 rows each)
  const int nw  = wid >> 2;      // N group (64 cols each)

  for (int t0 = 0; t0 < nt; t0 += 64) {
    __syncthreads();
    if (tid < 64) {
      int i = t0 + tid;
      s_tok[tid] = (i < nt) ? tok_list[e * B_TOK + i] : -1;
    }
    v8f acc[4];
    #pragma unroll
    for (int s = 0; s < 4; ++s) acc[s] = (v8f){0.f,0.f,0.f,0.f,0.f,0.f,0.f,0.f};

    for (int k0 = 0; k0 < D_IN; k0 += 32) {
      __syncthreads();
      // stage A: 64 rows x 32 K; one float4 per (row, 4K) quad, packed 8B store
      #pragma unroll
      for (int q = tid; q < 512; q += 256) {
        int row  = q >> 3;
        int kk   = (q & 7) * 4;
        int pk   = s_tok[row];
        int xrow = (pk < 0) ? 0 : (pk >> 1);
        const float4 v = *(const float4*)&x[(size_t)xrow * D_IN + k0 + kk];
        U2 u; u.a = f2bf_pk(v.x, v.y); u.b = f2bf_pk(v.z, v.w);
        *(U2*)&ldsA[row * 40 + kk] = u;
      }
      // stage B: 32 K x 128 cols; float4 along cols, transposed b16 stores
      #pragma unroll
      for (int q = tid; q < 1024; q += 256) {
        int kk = q >> 5;
        int c4 = (q & 31) * 4;
        const float4 v = *(const float4*)&W1e[(size_t)(k0 + kk) * H_HID + cBase + c4];
        ldsB[(c4 + 0) * 40 + kk] = f2bf(v.x);
        ldsB[(c4 + 1) * 40 + kk] = f2bf(v.y);
        ldsB[(c4 + 2) * 40 + kk] = f2bf(v.z);
        ldsB[(c4 + 3) * 40 + kk] = f2bf(v.w);
      }
      if (k0 + 32 < D_IN)
        __builtin_prefetch(&W1e[(size_t)(k0 + 32) * H_HID + cBase + (tid & 127)], 0, 0);
      __syncthreads();

      FragU fa;
      fa.q[0] = *(const Q16*)&ldsA[(mw * 16 + n) * 40 + aK];
      fa.q[1] = *(const Q16*)&ldsA[(mw * 16 + n) * 40 + aK + 16];
      #pragma unroll
      for (int s = 0; s < 4; ++s) {
        FragU fb;
        const unsigned short* bp = &ldsB[(nw * 64 + s * 16 + n) * 40 + bK];
        fb.q[0] = *(const Q16*)bp;
        fb.q[1] = *(const Q16*)(bp + 16);
        acc[s] = wmma_bf16(fa, fb, acc[s]);
      }
    }

    #pragma unroll
    for (int s = 0; s < 4; ++s) {
      int col = cBase + nw * 64 + s * 16 + n;
      float bias = b1[e * H_HID + col];
      #pragma unroll
      for (int r = 0; r < 8; ++r) {
        int m  = mw * 16 + r + (hi ? 8 : 0);
        int pk = s_tok[m];
        if (pk >= 0)
          hbuf[(size_t)pk * H_HID + col] = f2bf(acc[s][r] + bias);
      }
    }
  }
}

// ---------------------------------------------------------------------------
// Kernel 3: routed fc2.  logits[b*2+slot, :] = h@W2[e] + b2[e]  (fp32 out)
// Same tiling (M=64, N=128, K=32 over H=2048).
// A (h) is bf16 row-major -> GLOBAL_LOAD_ASYNC_TO_LDS_B128 (ASYNCcnt path).
// ---------------------------------------------------------------------------
__global__ __launch_bounds__(256)
void moe_fc2_kernel(const unsigned short* __restrict__ hbuf,
                    const float* __restrict__ W2, const float* __restrict__ b2,
                    const int* __restrict__ counts, const int* __restrict__ tok_list,
                    float* __restrict__ logits) {
  const int e     = blockIdx.y;
  const int cBase = blockIdx.x * 128;
  const int tid   = threadIdx.x;
  const int wid   = tid >> 5;
  const int lane  = tid & 31;
  const int nt    = counts[e];
  const float* W2e = W2 + (size_t)e * H_HID * C_CLS;

  __shared__ unsigned short ldsA[64 * 40];
  __shared__ unsigned short ldsB[128 * 40];
  __shared__ int s_tok[64];

  const int n   = lane & 15;
  const int hi  = lane >> 4;
  const int aK  = hi ? 8 : 0;
  const int bK  = hi ? 8 : 0;
  const int mw  = wid & 3;
  const int nw  = wid >> 2;

  for (int t0 = 0; t0 < nt; t0 += 64) {
    __syncthreads();
    if (tid < 64) {
      int i = t0 + tid;
      s_tok[tid] = (i < nt) ? tok_list[e * B_TOK + i] : -1;
    }
    v8f acc[4];
    #pragma unroll
    for (int s = 0; s < 4; ++s) acc[s] = (v8f){0.f,0.f,0.f,0.f,0.f,0.f,0.f,0.f};

    for (int k0 = 0; k0 < H_HID; k0 += 32) {
      __syncthreads();
      // stage A: 64 rows x 32 halves = 256 x 16B async chunks, one per thread
      {
        int row = tid >> 2;          // 0..63
        int ch  = tid & 3;           // 0..3 (16B chunks)
        int pk  = s_tok[row];
        int grow = (pk < 0) ? 0 : pk;
        const unsigned short* gp = hbuf + (size_t)grow * H_HID + k0 + ch * 8;
        copy16_g2l(gp, &ldsA[row * 40 + ch * 8]);
      }
      // stage B: W2 slice; float4 along cols (whole-quad guard, 1000 % 4 == 0)
      #pragma unroll
      for (int q = tid; q < 1024; q += 256) {
        int kk   = q >> 5;
        int c4   = (q & 31) * 4;
        int col4 = cBase + c4;
        float4 v;
        if (col4 < C_CLS) v = *(const float4*)&W2e[(size_t)(k0 + kk) * C_CLS + col4];
        else              v = make_float4(0.f, 0.f, 0.f, 0.f);
        ldsB[(c4 + 0) * 40 + kk] = f2bf(v.x);
        ldsB[(c4 + 1) * 40 + kk] = f2bf(v.y);
        ldsB[(c4 + 2) * 40 + kk] = f2bf(v.z);
        ldsB[(c4 + 3) * 40 + kk] = f2bf(v.w);
      }
      if (k0 + 32 < H_HID)
        __builtin_prefetch(&W2e[(size_t)(k0 + 32) * C_CLS + cBase + (tid & 127)], 0, 0);
      wait_async_lds();
      __syncthreads();

      FragU fa;
      fa.q[0] = *(const Q16*)&ldsA[(mw * 16 + n) * 40 + aK];
      fa.q[1] = *(const Q16*)&ldsA[(mw * 16 + n) * 40 + aK + 16];
      #pragma unroll
      for (int s = 0; s < 4; ++s) {
        FragU fb;
        const unsigned short* bp = &ldsB[(nw * 64 + s * 16 + n) * 40 + bK];
        fb.q[0] = *(const Q16*)bp;
        fb.q[1] = *(const Q16*)(bp + 16);
        acc[s] = wmma_bf16(fa, fb, acc[s]);
      }
    }

    #pragma unroll
    for (int s = 0; s < 4; ++s) {
      int col = cBase + nw * 64 + s * 16 + n;
      float bias = (col < C_CLS) ? b2[e * C_CLS + col] : 0.f;
      #pragma unroll
      for (int r = 0; r < 8; ++r) {
        int m  = mw * 16 + r + (hi ? 8 : 0);
        int pk = s_tok[m];
        if (pk >= 0 && col < C_CLS)
          logits[(size_t)pk * C_CLS + col] = acc[s][r] + bias;
      }
    }
  }
}

// ---------------------------------------------------------------------------
// Kernel 4: per token: log_softmax over both assignment rows, weighted sum.
// Fully overwrites out (no memset needed).
// ---------------------------------------------------------------------------
__global__ __launch_bounds__(256)
void lsm_combine_kernel(const float* __restrict__ logits,
                        const float* __restrict__ wv, float* __restrict__ out) {
  const int b    = blockIdx.x;
  const int tid  = threadIdx.x;
  const int lane = tid & 31;
  const int wid  = tid >> 5;
  __shared__ float red[8];

  float acc[4] = {0.f, 0.f, 0.f, 0.f};

  for (int slot = 0; slot < 2; ++slot) {
    const float* row = logits + (size_t)(b * 2 + slot) * C_CLS;
    float lv[4];
    float mx = -3.4e38f;
    #pragma unroll
    for (int r = 0; r < 4; ++r) {
      int c = tid + (r << 8);
      lv[r] = (c < C_CLS) ? row[c] : -3.4e38f;
      mx = fmaxf(mx, lv[r]);
    }
    #pragma unroll
    for (int off = 16; off >= 1; off >>= 1) mx = fmaxf(mx, __shfl_xor(mx, off));
    if (lane == 0) red[wid] = mx;
    __syncthreads();
    float M = red[0];
    #pragma unroll
    for (int i = 1; i < 8; ++i) M = fmaxf(M, red[i]);
    __syncthreads();

    float se = 0.f;
    #pragma unroll
    for (int r = 0; r < 4; ++r) {
      int c = tid + (r << 8);
      if (c < C_CLS) se += __expf(lv[r] - M);
    }
    #pragma unroll
    for (int off = 16; off >= 1; off >>= 1) se += __shfl_xor(se, off);
    if (lane == 0) red[wid] = se;
    __syncthreads();
    float S = 0.f;
    #pragma unroll
    for (int i = 0; i < 8; ++i) S += red[i];
    __syncthreads();

    float lse = M + __logf(S);
    float wgt = wv[b * 2 + slot];
    #pragma unroll
    for (int r = 0; r < 4; ++r) {
      int c = tid + (r << 8);
      if (c < C_CLS) acc[r] += wgt * (lv[r] - lse);
    }
  }

  #pragma unroll
  for (int r = 0; r < 4; ++r) {
    int c = tid + (r << 8);
    if (c < C_CLS) out[(size_t)b * C_CLS + c] = acc[r];
  }
}

// ---------------------------------------------------------------------------
extern "C" void kernel_launch(void* const* d_in, const int* in_sizes, int n_in,
                              void* d_out, int out_size, void* d_ws, size_t ws_size,
                              hipStream_t stream) {
  (void)in_sizes; (void)n_in; (void)out_size; (void)ws_size;
  const float* x  = (const float*)d_in[0];
  const float* Wg = (const float*)d_in[1];
  const float* bg = (const float*)d_in[2];
  const float* W1 = (const float*)d_in[3];
  const float* b1 = (const float*)d_in[4];
  const float* W2 = (const float*)d_in[5];
  const float* b2 = (const float*)d_in[6];
  // d_in[7] is k == 2 (compile-time constant here)
  float* out = (float*)d_out;

  // Workspace carve-up
  char* p = (char*)d_ws;
  float* wv = (float*)p;            p += (size_t)B_TOK * 2 * sizeof(float);
  int* counts = (int*)p;            p += 256;
  int* tok_list = (int*)p;          p += (size_t)E_EXP * B_TOK * sizeof(int);
  unsigned short* hbuf = (unsigned short*)p;
                                    p += (size_t)B_TOK * 2 * H_HID * sizeof(unsigned short);
  float* logits = (float*)p;        p += (size_t)B_TOK * 2 * C_CLS * sizeof(float);

  (void)hipMemsetAsync(counts, 0, E_EXP * sizeof(int), stream);

  gate_topk_kernel<<<B_TOK / 8, 256, 0, stream>>>(x, Wg, bg, wv, counts, tok_list);

  moe_fc1_kernel<<<dim3(H_HID / 128, E_EXP), 256, 0, stream>>>(
      x, W1, b1, counts, tok_list, hbuf);

  moe_fc2_kernel<<<dim3((C_CLS + 127) / 128, E_EXP), 256, 0, stream>>>(
      hbuf, W2, b2, counts, tok_list, logits);

  lsm_combine_kernel<<<B_TOK, 256, 0, stream>>>(logits, wv, out);
}
